// TransformerLayer_83356725281355
// MI455X (gfx1250) — matrix operand order repaired and checked by reading, not compiled, not run on hardware
//
#include <hip/hip_runtime.h>
#include <math.h>

// ---------------------------------------------------------------------------
// Problem constants (match reference)
// ---------------------------------------------------------------------------
#define B_  2
#define S_  2048
#define D_  2048
#define H_  16
#define HD_ 128
#define FF_ 8192
#define M_  (B_ * S_)   // 4096 total rows

typedef __bf16 bf16;
typedef __bf16 v16bf __attribute__((ext_vector_type(16)));
typedef __bf16 bf16x8 __attribute__((ext_vector_type(8)));
typedef float  v8f    __attribute__((ext_vector_type(8)));
typedef int    v4i    __attribute__((ext_vector_type(4)));

__device__ __forceinline__ v8f wmma_bf16(v16bf a, v16bf b, v8f c) {
  // D = A(16x32 bf16) * B(32x16 bf16) + C(16x16 f32)
  return __builtin_amdgcn_wmma_f32_16x16x32_bf16(
      /*neg_a=*/false, a, /*neg_b=*/false, b,
      /*c_mod=*/(short)0, c, /*reuse_a=*/false, /*reuse_b=*/false);
}

// ---------------------------------------------------------------------------
// CDNA5 async memory->LDS copy (ASYNCcnt-tracked), with safe fallback.
// Probe-confirmed: param0 is v4i addrspace(1)*, so pass typed AS pointers.
// ---------------------------------------------------------------------------
#if __has_builtin(__builtin_amdgcn_global_load_async_to_lds_b128)
#define USE_ASYNC_LDS 1
#else
#define USE_ASYNC_LDS 0
#endif

#if USE_ASYNC_LDS
__device__ __forceinline__ void async_copy16(const void* g, void* l) {
  __builtin_amdgcn_global_load_async_to_lds_b128(
      (__attribute__((address_space(1))) v4i*)g,
      (__attribute__((address_space(3))) v4i*)l,
      /*imm offset=*/0, /*cpol=*/0);
}
__device__ __forceinline__ void wait_async0() {
#if __has_builtin(__builtin_amdgcn_s_wait_asynccnt)
  __builtin_amdgcn_s_wait_asynccnt(0);
#else
  asm volatile("s_wait_asynccnt 0x0" ::: "memory");
#endif
}
#endif

// ---------------------------------------------------------------------------
// fp32 -> bf16 conversion (weights / one-time)
// ---------------------------------------------------------------------------
__global__ void k_f2bf(const float* __restrict__ in, bf16* __restrict__ out, int n) {
  int i = blockIdx.x * blockDim.x + threadIdx.x;
  int stride = gridDim.x * blockDim.x;
  for (; i < n; i += stride) out[i] = (bf16)in[i];
}

// ---------------------------------------------------------------------------
// LayerNorm over D=2048, one block (256 threads) per row, bf16 output
// ---------------------------------------------------------------------------
__global__ __launch_bounds__(256)
void k_layernorm(const float* __restrict__ x, const float* __restrict__ g,
                 const float* __restrict__ b, bf16* __restrict__ out) {
  __shared__ float red[256];
  const int row = blockIdx.x;
  const float* xr = x + (size_t)row * D_;

  float s = 0.f;
  for (int i = threadIdx.x; i < D_; i += 256) s += xr[i];
  red[threadIdx.x] = s;
  __syncthreads();
  for (int off = 128; off > 0; off >>= 1) {
    if (threadIdx.x < off) red[threadIdx.x] += red[threadIdx.x + off];
    __syncthreads();
  }
  const float mu = red[0] * (1.0f / D_);
  __syncthreads();

  float v = 0.f;
  for (int i = threadIdx.x; i < D_; i += 256) {
    float d = xr[i] - mu;
    v += d * d;
  }
  red[threadIdx.x] = v;
  __syncthreads();
  for (int off = 128; off > 0; off >>= 1) {
    if (threadIdx.x < off) red[threadIdx.x] += red[threadIdx.x + off];
    __syncthreads();
  }
  const float rstd = rsqrtf(red[0] * (1.0f / D_) + 1e-5f);

  bf16* orow = out + (size_t)row * D_;
  for (int i = threadIdx.x; i < D_; i += 256)
    orow[i] = (bf16)((xr[i] - mu) * rstd * g[i] + b[i]);
}

// ---------------------------------------------------------------------------
// Tiled bf16 GEMM with WMMA:  out[M,N] = A[M,K] * Bw[K,N] + bias (+gelu)(+resid)
// Block: 256 threads = 8 waves; tile 128x64; each wave 32x32 (2x2 wmma frags).
// Staging uses CDNA5 async global->LDS copies when available, plus
// global_prefetch of the next K chunk.
// ---------------------------------------------------------------------------
template <bool GELU, bool RESID, bool OUTBF>
__global__ __launch_bounds__(256)
void k_gemm(const bf16* __restrict__ A, const bf16* __restrict__ Bw,
            const float* __restrict__ bias, const float* __restrict__ resid,
            void* __restrict__ outp, int M, int N, int K) {
  __shared__ __align__(16) bf16 As[128][40];  // 32 cols + pad (80B rows, 16B aligned)
  __shared__ __align__(16) bf16 Bs[32][72];   // 64 cols + pad (144B rows)

  const int tid  = threadIdx.x;
  const int lane = tid & 31;
  const int wid  = tid >> 5;
  const int wm   = wid & 3;       // wave row  0..3  (32 rows each)
  const int wn   = wid >> 2;      // wave col  0..1  (32 cols each)
  const int n16  = lane & 15;
  const int hi   = lane >> 4;

  const int bm = blockIdx.y * 128;
  const int bn = blockIdx.x * 64;

  // staging assignments
  const int arow = tid >> 2, acol = (tid & 3) * 8;  // A: 128x32, 2 vectors/thread
  const int brow = tid >> 3, bcol = (tid & 7) * 8;  // B: 32x64,  1 vector/thread

  v8f acc[2][2] = {};

  for (int k0 = 0; k0 < K; k0 += 32) {
    const bf16* ga0 = A + (size_t)(bm + arow) * K + k0 + acol;
    const bf16* ga1 = A + (size_t)(bm + arow + 64) * K + k0 + acol;
    const bf16* gb  = Bw + (size_t)(k0 + brow) * N + bn + bcol;

    __syncthreads();
#if USE_ASYNC_LDS
    async_copy16(ga0, &As[arow][acol]);
    async_copy16(ga1, &As[arow + 64][acol]);
    async_copy16(gb, &Bs[brow][bcol]);
#else
    *(bf16x8*)&As[arow][acol]      = *(const bf16x8*)ga0;
    *(bf16x8*)&As[arow + 64][acol] = *(const bf16x8*)ga1;
    *(bf16x8*)&Bs[brow][bcol]      = *(const bf16x8*)gb;
#endif
    // prefetch next K chunk into cache while WMMAs run
    if (k0 + 32 < K) {
      __builtin_prefetch(ga0 + 32, 0, 1);
      __builtin_prefetch(gb + (size_t)32 * N, 0, 1);
    }
#if USE_ASYNC_LDS
    wait_async0();  // drain this wave's ASYNCcnt before the barrier
#endif
    __syncthreads();

    // A fragments (16x32): rows across lanes 0-15, K pairs across VGPRs
    v16bf af[2];
#pragma unroll
    for (int i = 0; i < 2; ++i) {
      const int ra = wm * 32 + i * 16 + n16;
#pragma unroll
      for (int v = 0; v < 8; ++v) {
        const int kb = 2 * v + 16 * hi;
        union { unsigned u; bf16 h[2]; } t;
        t.u = *(const unsigned*)&As[ra][kb];
        af[i][2 * v] = t.h[0];
        af[i][2 * v + 1] = t.h[1];
      }
    }
    // B fragments (32x16): N across lanes, K pairs across VGPRs (hi half: K+16)
    v16bf bfr[2];
#pragma unroll
    for (int j = 0; j < 2; ++j) {
      const int cb = wn * 32 + j * 16 + n16;
#pragma unroll
      for (int v = 0; v < 8; ++v) {
        const int kb = 2 * v + 16 * hi;
        bfr[j][2 * v] = Bs[kb][cb];
        bfr[j][2 * v + 1] = Bs[kb + 1][cb];
      }
    }
#pragma unroll
    for (int i = 0; i < 2; ++i)
#pragma unroll
      for (int j = 0; j < 2; ++j)
        acc[i][j] = wmma_bf16(af[i], bfr[j], acc[i][j]);
  }

  // epilogue: C layout -> row = base + r + 8*hi, col = base + (lane&15)
#pragma unroll
  for (int i = 0; i < 2; ++i) {
#pragma unroll
    for (int j = 0; j < 2; ++j) {
      const int col = bn + wn * 32 + j * 16 + n16;
#pragma unroll
      for (int r = 0; r < 8; ++r) {
        const int row = bm + wm * 32 + i * 16 + r + 8 * hi;
        float val = acc[i][j][r] + bias[col];
        if (GELU) val = 0.5f * val * (1.0f + erff(val * 0.70710678118654752f));
        if (RESID) val += resid[(size_t)row * N + col];
        if (OUTBF)
          ((bf16*)outp)[(size_t)row * N + col] = (bf16)val;
        else
          ((float*)outp)[(size_t)row * N + col] = val;
      }
    }
  }
}

// ---------------------------------------------------------------------------
// Flash attention (causal, MQA: one KV head). One wave per 16 q-rows.
// q:[B,S,H,HD] bf16, k/v:[B,S,HD] bf16, o:[B,S,H,HD] bf16
// ---------------------------------------------------------------------------
__global__ __launch_bounds__(32)
void k_attn(const bf16* __restrict__ q, const bf16* __restrict__ kmat,
            const bf16* __restrict__ vmat, bf16* __restrict__ o) {
  __shared__ __align__(16) bf16 pt[16][32];  // P tile for C->A transpose

  const int lane = threadIdx.x & 31;
  const int n16 = lane & 15;
  const int hi  = lane >> 4;
  const int b   = blockIdx.z;
  const int h   = blockIdx.y;
  const int q0  = blockIdx.x * 16;
  const float scale = 0.08838834764831845f;  // 1/sqrt(128)

  // Preload Q as 4 A-fragments (HD=128 = 4 chunks of K=32)
  v16bf qf[4];
  const bf16* qbase = q + (((size_t)(b * S_ + q0 + n16) * H_) + h) * HD_;
#pragma unroll
  for (int c = 0; c < 4; ++c) {
#pragma unroll
    for (int v = 0; v < 8; ++v) {
      const int kb = 2 * v + 16 * hi;
      union { unsigned u; bf16 h2[2]; } t;
      t.u = *(const unsigned*)(qbase + c * 32 + kb);
      qf[c][2 * v] = t.h2[0];
      qf[c][2 * v + 1] = t.h2[1];
    }
  }

  v8f oacc[8] = {};
  float mrow[8], lrow[8];
#pragma unroll
  for (int r = 0; r < 8; ++r) { mrow[r] = -1e30f; lrow[r] = 0.f; }

  const int ntiles = (q0 + 16 + 31) / 32;  // causal bound (kb0 <= q0 always)
  for (int kt = 0; kt < ntiles; ++kt) {
    const int kb0 = kt * 32;

    // S = Q * K^T over this 32-wide KV tile: two 16x16 C fragments
    v8f s0 = {}, s1 = {};
#pragma unroll
    for (int c = 0; c < 4; ++c) {
      v16bf kf0, kf1;
#pragma unroll
      for (int v = 0; v < 8; ++v) {
        const int kp = 2 * v + 16 * hi;  // contraction = hd index within chunk
        union { unsigned u; bf16 h2[2]; } t0, t1;
        t0.u = *(const unsigned*)(kmat + (size_t)(b * S_ + kb0 + n16) * HD_ + c * 32 + kp);
        t1.u = *(const unsigned*)(kmat + (size_t)(b * S_ + kb0 + 16 + n16) * HD_ + c * 32 + kp);
        kf0[2 * v] = t0.h2[0]; kf0[2 * v + 1] = t0.h2[1];
        kf1[2 * v] = t1.h2[0]; kf1[2 * v + 1] = t1.h2[1];
      }
      s0 = wmma_bf16(qf[c], kf0, s0);
      s1 = wmma_bf16(qf[c], kf1, s1);
    }

    // causal mask + online softmax (row = r + 8*hi across 16-lane halves)
#pragma unroll
    for (int r = 0; r < 8; ++r) {
      const int qrow = q0 + r + 8 * hi;
      float a0 = s0[r] * scale;
      float a1 = s1[r] * scale;
      if (kb0 + n16 > qrow) a0 = -1e30f;
      if (kb0 + 16 + n16 > qrow) a1 = -1e30f;
      float mx = fmaxf(a0, a1);
#pragma unroll
      for (int off = 1; off < 16; off <<= 1) mx = fmaxf(mx, __shfl_xor(mx, off, 32));
      const float mnew = fmaxf(mrow[r], mx);
      const float alpha = __expf(mrow[r] - mnew);
      const float p0 = __expf(a0 - mnew);
      const float p1 = __expf(a1 - mnew);
      float rs = p0 + p1;
#pragma unroll
      for (int off = 1; off < 16; off <<= 1) rs += __shfl_xor(rs, off, 32);
      lrow[r] = lrow[r] * alpha + rs;
      mrow[r] = mnew;
#pragma unroll
      for (int j = 0; j < 8; ++j) oacc[j][r] *= alpha;
      pt[r + 8 * hi][n16] = (bf16)p0;
      pt[r + 8 * hi][n16 + 16] = (bf16)p1;
    }
    __syncthreads();  // single wave: orders LDS stores before A-frag loads

    // P (16x32) as A fragment via LDS transpose
    v16bf pA;
#pragma unroll
    for (int v = 0; v < 8; ++v) {
      const int kb = 2 * v + 16 * hi;
      union { unsigned u; bf16 h2[2]; } t;
      t.u = *(const unsigned*)&pt[n16][kb];
      pA[2 * v] = t.h2[0];
      pA[2 * v + 1] = t.h2[1];
    }

    // O += P * V  (V tile 32x128 -> 8 B-fragments across hd)
#pragma unroll
    for (int j = 0; j < 8; ++j) {
      v16bf vf;
#pragma unroll
      for (int v = 0; v < 8; ++v) {
        const int kp = 2 * v + 16 * hi;  // contraction = kv index (strided)
        vf[2 * v]     = vmat[(size_t)(b * S_ + kb0 + kp) * HD_ + j * 16 + n16];
        vf[2 * v + 1] = vmat[(size_t)(b * S_ + kb0 + kp + 1) * HD_ + j * 16 + n16];
      }
      oacc[j] = wmma_bf16(pA, vf, oacc[j]);
    }
    __syncthreads();
  }

  // Normalize and write O
#pragma unroll
  for (int j = 0; j < 8; ++j) {
#pragma unroll
    for (int r = 0; r < 8; ++r) {
      const int row = q0 + r + 8 * hi;
      const float val = oacc[j][r] / lrow[r];
      o[(((size_t)(b * S_ + row) * H_) + h) * HD_ + j * 16 + n16] = (bf16)val;
    }
  }
}

// ---------------------------------------------------------------------------
// Host-side orchestration
// ---------------------------------------------------------------------------
extern "C" void kernel_launch(void* const* d_in, const int* in_sizes, int n_in,
                              void* d_out, int out_size, void* d_ws, size_t ws_size,
                              hipStream_t stream) {
  (void)in_sizes; (void)n_in; (void)out_size; (void)ws_size;

  const float* x     = (const float*)d_in[0];
  // d_in[1] attn_bias: unused (causal mask applied analytically)
  const float* wq    = (const float*)d_in[2];
  const float* wq_b  = (const float*)d_in[3];
  const float* wk    = (const float*)d_in[4];
  const float* wk_b  = (const float*)d_in[5];
  const float* wv    = (const float*)d_in[6];
  const float* wv_b  = (const float*)d_in[7];
  const float* wo    = (const float*)d_in[8];
  const float* wo_b  = (const float*)d_in[9];
  const float* w1    = (const float*)d_in[10];
  const float* w1_b  = (const float*)d_in[11];
  const float* w2    = (const float*)d_in[12];
  const float* w2_b  = (const float*)d_in[13];
  const float* ln1_s = (const float*)d_in[14];
  const float* ln1_b = (const float*)d_in[15];
  const float* ln2_s = (const float*)d_in[16];
  const float* ln2_b = (const float*)d_in[17];

  char* ws = (char*)d_ws;
  size_t off = 0;
  auto alloc = [&](size_t bytes) {
    void* p = ws + off;
    off = (off + bytes + 255) & ~(size_t)255;
    return p;
  };

  bf16* hb   = (bf16*)alloc((size_t)M_ * D_ * 2);   // LN1 output
  bf16* h2b  = (bf16*)alloc((size_t)M_ * D_ * 2);   // LN2 output
  bf16* wqb  = (bf16*)alloc((size_t)D_ * D_ * 2);
  bf16* wkb  = (bf16*)alloc((size_t)D_ * HD_ * 2);
  bf16* wvb  = (bf16*)alloc((size_t)D_ * HD_ * 2);
  bf16* wob  = (bf16*)alloc((size_t)D_ * D_ * 2);
  bf16* w1b  = (bf16*)alloc((size_t)D_ * FF_ * 2);
  bf16* w2b  = (bf16*)alloc((size_t)FF_ * D_ * 2);
  bf16* qb   = (bf16*)alloc((size_t)M_ * D_ * 2);   // [B,S,H,HD]
  bf16* kb   = (bf16*)alloc((size_t)M_ * HD_ * 2);  // [B,S,HD]
  bf16* vb   = (bf16*)alloc((size_t)M_ * HD_ * 2);
  bf16* ob   = (bf16*)alloc((size_t)M_ * D_ * 2);   // attention output
  float* x2  = (float*)alloc((size_t)M_ * D_ * 4);  // x + attn_out
  bf16* ff1b = (bf16*)alloc((size_t)M_ * FF_ * 2);  // gelu(h2@w1+b1)

  // 1) weights fp32 -> bf16
  k_f2bf<<<2048, 256, 0, stream>>>(wq, wqb, D_ * D_);
  k_f2bf<<<2048, 256, 0, stream>>>(wk, wkb, D_ * HD_);
  k_f2bf<<<2048, 256, 0, stream>>>(wv, wvb, D_ * HD_);
  k_f2bf<<<2048, 256, 0, stream>>>(wo, wob, D_ * D_);
  k_f2bf<<<2048, 256, 0, stream>>>(w1, w1b, D_ * FF_);
  k_f2bf<<<2048, 256, 0, stream>>>(w2, w2b, FF_ * D_);

  // 2) LN1
  k_layernorm<<<M_, 256, 0, stream>>>(x, ln1_s, ln1_b, hb);

  // 3) Q / K / V projections (bf16 out, bias)
  k_gemm<false, false, true><<<dim3(D_ / 64, M_ / 128), 256, 0, stream>>>(
      hb, wqb, wq_b, nullptr, qb, M_, D_, D_);
  k_gemm<false, false, true><<<dim3(HD_ / 64, M_ / 128), 256, 0, stream>>>(
      hb, wkb, wk_b, nullptr, kb, M_, HD_, D_);
  k_gemm<false, false, true><<<dim3(HD_ / 64, M_ / 128), 256, 0, stream>>>(
      hb, wvb, wv_b, nullptr, vb, M_, HD_, D_);

  // 4) causal flash attention
  k_attn<<<dim3(S_ / 16, H_, B_), 32, 0, stream>>>(qb, kb, vb, ob);

  // 5) output projection + residual  -> x2 (fp32)
  k_gemm<false, true, false><<<dim3(D_ / 64, M_ / 128), 256, 0, stream>>>(
      ob, wob, wo_b, x, x2, M_, D_, D_);

  // 6) LN2
  k_layernorm<<<M_, 256, 0, stream>>>(x2, ln2_s, ln2_b, h2b);

  // 7) FFN1 + exact GELU (bf16 out)
  k_gemm<true, false, true><<<dim3(FF_ / 64, M_ / 128), 256, 0, stream>>>(
      h2b, w1b, w1_b, nullptr, ff1b, M_, FF_, D_);

  // 8) FFN2 + residual -> d_out (fp32)
  k_gemm<false, true, false><<<dim3(D_ / 64, M_ / 128), 256, 0, stream>>>(
      ff1b, w2b, w2_b, x2, (float*)d_out, M_, D_, FF_);
}
